// PredDecoder_41120016892049
// MI455X (gfx1250) — matrix operand order repaired and checked by reading, not compiled, not run on hardware
//
#include <hip/hip_runtime.h>
#include <hip/hip_bf16.h>

typedef __bf16 bf16_t;
typedef __bf16 bf16x8 __attribute__((ext_vector_type(8)));
typedef __bf16 v16bf  __attribute__((ext_vector_type(16)));
typedef float  v8f    __attribute__((ext_vector_type(8)));

#define T_LEN 1024
#define I_SZ  128
#define H_SZ  512
#define G4H   2048

// Packed weight layouts (built by prepass):
//   Wg: [ntg(128)][kt(20)][lane(32)][e(16)] bf16   (kt 0..15 = W_hh, 16..19 = W_ih)
//   Wo: [nt(8)][kt(16)][lane(32)][e(16)]   bf16
// B-fragment (32x16 bf16, wave32): lane l, elem e -> K = (e>=8?16:0)+(l>=16?8:0)+(e&7), N = l&15
#define WG_ELEMS (128 * 20 * 512)
#define WO_ELEMS (8 * 16 * 512)

__device__ __forceinline__ float sigm(float v) {
    return 1.0f / (1.0f + __expf(-v));
}

// Fast tanh: native v_tanh_f32 when the builtin exists, else via native exp.
__device__ __forceinline__ float fast_tanh(float v) {
#if __has_builtin(__builtin_amdgcn_tanhf)
    return __builtin_amdgcn_tanhf(v);
#else
    return 2.0f / (1.0f + __expf(-2.0f * v)) - 1.0f;
#endif
}

// A-fragment (16x32 bf16, wave32): lane l (m=l&15), elem e ->
//   K = (e>=8?16:0) + ((l>=16)?8:0) + (e&7)
// So per lane: two contiguous 16-byte LDS loads.
__device__ __forceinline__ v16bf load_a_frag(const bf16_t* base, int stride,
                                             int kbase, int lane) {
    const int m    = lane & 15;
    const int koff = (lane >> 4) << 3; // 0 or 8
    const bf16_t* p = base + m * stride + kbase + koff;
    union { v16bf v; bf16x8 h[2]; } u;
    u.h[0] = *(const bf16x8*)(p);
    u.h[1] = *(const bf16x8*)(p + 16);
    return u.v;
}

// ---------------- prepass: pack weights to bf16 fragment layout ----------------
__global__ void pack_wg(const float* __restrict__ W_hh,
                        const float* __restrict__ W_ih,
                        bf16_t* __restrict__ Wg) {
    int idx  = blockIdx.x * 256 + threadIdx.x;
    int e    = idx & 15;
    int lane = (idx >> 4) & 31;
    int tile = idx >> 9;          // ntg*20 + kt
    int kt   = tile % 20;
    int ntg  = tile / 20;
    int n    = ntg * 16 + (lane & 15);
    int kl   = ((e >> 3) << 4) + ((lane >> 4) << 3) + (e & 7);   // 0..31 within k-tile
    float v;
    if (kt < 16) v = W_hh[n * H_SZ + kt * 32 + kl];
    else         v = W_ih[n * I_SZ + (kt - 16) * 32 + kl];
    Wg[idx] = (bf16_t)v;
}

__global__ void pack_wo(const float* __restrict__ W_out, bf16_t* __restrict__ Wo) {
    int idx  = blockIdx.x * 256 + threadIdx.x;
    int e    = idx & 15;
    int lane = (idx >> 4) & 31;
    int tile = idx >> 9;          // nt*16 + kt
    int kt   = tile & 15;
    int nt   = tile >> 4;
    int n    = nt * 16 + (lane & 15);
    int kl   = ((e >> 3) << 4) + ((lane >> 4) << 3) + (e & 7);
    Wo[idx] = (bf16_t)W_out[n * H_SZ + kt * 32 + kl];
}

__global__ void pack_bias(const float* __restrict__ b_ih,
                          const float* __restrict__ b_hh,
                          float* __restrict__ biasg) {
    int i = blockIdx.x * 256 + threadIdx.x;
    biasg[i] = b_ih[i] + b_hh[i];
}

// ---------------- fused persistent LSTM + projection ----------------
// grid = 8 blocks (16 batch rows each), block = 1024 threads = 32 waves.
// wave w owns hidden columns [16w, 16w+16): computes its own i/f/g/o tiles
// (4 accumulators = 32 VGPRs), keeps c in C-fragment registers, writes h
// (bf16) back to LDS. 8 waves/SIMD hides the L2-resident weight-load latency.
// All weight-fragment loops are kept as *loops* (no full unroll) so their
// time-invariant addresses cannot be LICM'd out of the t-loop into spills.
__global__ __launch_bounds__(1024)
void lstm_fused(const float* __restrict__ x,
                const float* __restrict__ h0,
                const float* __restrict__ c0,
                const bf16_t* __restrict__ Wg,
                const bf16_t* __restrict__ Wo,
                const float* __restrict__ biasg,
                const float* __restrict__ b_out,
                float* __restrict__ out) {
    __shared__ bf16_t hbuf[2][16][520];   // padded stride: 520 -> bank-spread
    __shared__ bf16_t xbuf[16][136];

    const int tid  = threadIdx.x;
    const int lane = tid & 31;
    const int w    = tid >> 5;           // wave 0..31
    const int b0   = blockIdx.x * 16;
    const int mlo  = lane & 15;          // N within tile / M low for A
    const int mhi  = (lane >> 4) << 3;   // C-frag: m = r + mhi
    const int col0 = w * 16;             // this wave's hidden-column base

    // init h LDS (t=0 reads buffer 0)
    for (int idx = tid; idx < 16 * H_SZ; idx += 1024) {
        int r = idx >> 9, c = idx & 511;
        hbuf[0][r][c] = (bf16_t)h0[(b0 + r) * H_SZ + c];
    }
    // c state in C-fragment registers: cfrag[r] = c[m = r+mhi][col0 + mlo]
    float cfrag[8];
#pragma unroll
    for (int r = 0; r < 8; ++r)
        cfrag[r] = c0[(b0 + r + mhi) * H_SZ + col0 + mlo];

    // per-lane gate biases (constant over time)
    float bg[4];
#pragma unroll
    for (int G = 0; G < 4; ++G)
        bg[G] = biasg[G * H_SZ + col0 + mlo];
    const float bo = (w < 8) ? b_out[w * 16 + mlo] : 0.0f;

    // preload x_0
    for (int idx = tid; idx < 16 * I_SZ; idx += 1024) {
        int r = idx >> 7, c = idx & 127;
        xbuf[r][c] = (bf16_t)x[(size_t)(b0 + r) * T_LEN * I_SZ + c];
    }
    __syncthreads();

#pragma unroll 1
    for (int t = 0; t < T_LEN; ++t) {
        const int cur = t & 1, nxt = cur ^ 1;

        // ---- gates = [h | x_t] x [W_hh | W_ih]^T, 4 tiles (i/f/g/o) ----
        v8f acc[4] = {};
#pragma unroll 2
        for (int kt = 0; kt < 16; ++kt) {
            v16bf a = load_a_frag(&hbuf[cur][0][0], 520, kt * 32, lane);
            v16bf bm[4];
#pragma unroll
            for (int G = 0; G < 4; ++G)
                bm[G] = *(const v16bf*)(Wg + ((size_t)((G * 32 + w) * 20 + kt) << 9) + (lane << 4));
#pragma unroll
            for (int G = 0; G < 4; ++G)
                acc[G] = __builtin_amdgcn_wmma_f32_16x16x32_bf16(
                    false, a, false, bm[G], (short)0, acc[G], false, false);
        }
        // x-contribution: MUST stay a rolled loop — fully unrolling makes the
        // (time-invariant) weight addresses constant and LLVM hoists the loaded
        // fragments out of the t-loop, spilling 128 VGPRs to scratch.
#pragma unroll 1
        for (int kt = 0; kt < 4; ++kt) {
            v16bf a = load_a_frag(&xbuf[0][0], 136, kt * 32, lane);
            v16bf bm[4];
#pragma unroll
            for (int G = 0; G < 4; ++G)
                bm[G] = *(const v16bf*)(Wg + ((size_t)((G * 32 + w) * 20 + 16 + kt) << 9) + (lane << 4));
#pragma unroll
            for (int G = 0; G < 4; ++G)
                acc[G] = __builtin_amdgcn_wmma_f32_16x16x32_bf16(
                    false, a, false, bm[G], (short)0, acc[G], false, false);
        }

        // ---- activations + state update, entirely in registers ----
#pragma unroll
        for (int r = 0; r < 8; ++r) {
            float iv = sigm(acc[0][r] + bg[0]);
            float fv = sigm(acc[1][r] + bg[1]);
            float gv = fast_tanh(acc[2][r] + bg[2]);
            float ov = sigm(acc[3][r] + bg[3]);
            float cv = fv * cfrag[r] + iv * gv;
            cfrag[r] = cv;
            float hv = ov * fast_tanh(cv);
            hbuf[nxt][r + mhi][col0 + mlo] = (bf16_t)hv;
        }
        __syncthreads();   // h_{t+1} visible to all waves

        // ---- per-step projection out_t = h_{t+1} W_out^T + b_out (waves 0..7) ----
        if (w < 8) {
            v8f pacc = {};
#pragma unroll 2
            for (int kt = 0; kt < 16; ++kt) {
                v16bf a  = load_a_frag(&hbuf[nxt][0][0], 520, kt * 32, lane);
                v16bf bm = *(const v16bf*)(Wo + ((size_t)(w * 16 + kt) << 9) + (lane << 4));
                pacc = __builtin_amdgcn_wmma_f32_16x16x32_bf16(
                    false, a, false, bm, (short)0, pacc, false, false);
            }
#pragma unroll
            for (int r = 0; r < 8; ++r)
                out[(size_t)(b0 + r + mhi) * T_LEN * I_SZ + (size_t)t * I_SZ + w * 16 + mlo] =
                    pacc[r] + bo;
        }

        // stage x_{t+1}; prefetch x_{t+2} toward L2
        if (t + 1 < T_LEN) {
            for (int idx = tid; idx < 16 * I_SZ; idx += 1024) {
                int r = idx >> 7, c = idx & 127;
                xbuf[r][c] = (bf16_t)x[(size_t)(b0 + r) * T_LEN * I_SZ + (size_t)(t + 1) * I_SZ + c];
            }
            if (t + 2 < T_LEN && tid < 16)
                __builtin_prefetch(&x[(size_t)(b0 + tid) * T_LEN * I_SZ + (size_t)(t + 2) * I_SZ], 0, 1);
        }
        __syncthreads();
    }
}

extern "C" void kernel_launch(void* const* d_in, const int* in_sizes, int n_in,
                              void* d_out, int out_size, void* d_ws, size_t ws_size,
                              hipStream_t stream) {
    const float* x     = (const float*)d_in[0];
    const float* h0    = (const float*)d_in[1];
    const float* c0    = (const float*)d_in[2];
    const float* W_ih  = (const float*)d_in[3];
    const float* W_hh  = (const float*)d_in[4];
    const float* b_ih  = (const float*)d_in[5];
    const float* b_hh  = (const float*)d_in[6];
    const float* W_out = (const float*)d_in[7];
    const float* b_out = (const float*)d_in[8];
    float* out = (float*)d_out;

    char*   ws    = (char*)d_ws;
    bf16_t* Wg    = (bf16_t*)ws;
    bf16_t* Wo    = (bf16_t*)(ws + (size_t)WG_ELEMS * 2);
    float*  biasg = (float*)(ws + (size_t)WG_ELEMS * 2 + (size_t)WO_ELEMS * 2);

    pack_wg  <<<WG_ELEMS / 256, 256, 0, stream>>>(W_hh, W_ih, Wg);
    pack_wo  <<<WO_ELEMS / 256, 256, 0, stream>>>(W_out, Wo);
    pack_bias<<<G4H / 256,      256, 0, stream>>>(b_ih, b_hh, biasg);

    lstm_fused<<<8, 1024, 0, stream>>>(x, h0, c0, Wg, Wo, biasg, b_out, out);
}
